// SrnmSpmm_9878424780931
// MI455X (gfx1250) — matrix operand order repaired and checked by reading, not compiled, not run on hardware
//
#include <hip/hip_runtime.h>

// ---------------------------------------------------------------------------
// Sparse (2:8 column-gather) SpMM as a bf16 WMMA GEMM for gfx1250 (MI455X).
//   out[m, n] = sum_k x[m, keep[k]] * values[n, k] + bias[n]
//   keep[k] = (k>>1)*8 + (k&1)*2   (static V:N:M pattern)
// M = B*S = 16384, N = D_OUT = 4096, K = K_KEPT = 1024 (fp32 in/out).
//
// Round 4 change: pin occupancy to 4 waves/SIMD (amdgpu_waves_per_eu(4,4)).
// The default heuristic targeted 8 waves/SIMD => 128-VGPR budget, which
// forced the allocator to recycle one B-fragment octet and expose three
// s_wait_dscnt 0x0 LDS round-trips per K-step inside the WMMA stream.
// 4 waves/SIMD (the floor for an 8-wave block in CU mode anyway) doubles the
// budget so acc(64) + A-frags(16) + B-frags(32) + staging(~32) all stay live
// and the 12 ds_load_b128s pipeline ahead of 8 back-to-back WMMAs.
// ---------------------------------------------------------------------------

typedef __attribute__((ext_vector_type(16))) __bf16 v16bf;
typedef __attribute__((ext_vector_type(8)))  float  v8f;

#define D_IN   4096
#define D_OUT  4096
#define K_KEPT 1024
#define M_TOT  16384
#define BM     128
#define BN     128
#define BK     32
#define NSTEP  (K_KEPT / BK)   // 32
#define LDA    40              // ushort stride: 80B = 5*16B (aligned, bank-spread)

union Frag {
    uint4 q[2];
    v16bf v;
};

// Pack two f32 into packed bf16 {lo=a, hi=b}: round-half-up then one v_perm.
__device__ __forceinline__ unsigned int pack2_bf16(float a, float b) {
    unsigned int ua = __float_as_uint(a) + 0x8000u;
    unsigned int ub = __float_as_uint(b) + 0x8000u;
    // D bytes = [ua.b2, ua.b3, ub.b2, ub.b3]  (S0=ub -> idx 4..7, S1=ua -> 0..3)
    return __builtin_amdgcn_perm(ub, ua, 0x07060302u);
}

__global__ __launch_bounds__(256)
__attribute__((amdgpu_waves_per_eu(4, 4)))
void srnm_spmm_wmma_bf16(const float* __restrict__ x,
                         const float* __restrict__ values,
                         const float* __restrict__ bias,
                         float* __restrict__ out) {
    __shared__ __align__(16) unsigned short sA[2][BM * LDA];
    __shared__ __align__(16) unsigned short sB[2][BN * LDA];

    const int t  = threadIdx.x;
    const int m0 = blockIdx.y * BM;
    const int n0 = blockIdx.x * BN;

    // ---- staging role: 2 threads per row, each covers 16 of the 32 K slots
    const int sr = t >> 1;       // row 0..127 within tile
    const int sh = t & 1;        // k-half 0/1
    const float* xrow = x      + (size_t)(m0 + sr) * D_IN;
    const float* vrow = values + (size_t)(n0 + sr) * K_KEPT;

    // ---- compute role: wave grid 4(M) x 2(N), each wave 32x64 outputs
    const int lane = t & 31;
    const int wave = t >> 5;
    const int wm   = wave & 3;            // 0..3 -> 32-row slots
    const int wn   = wave >> 2;           // 0..1 -> 64-col slots
    const int lrow = lane & 15;
    const int lhi  = (lane >> 4) & 1;     // lane half selects K sub-slab

    v8f acc[2][4];
#pragma unroll
    for (int i = 0; i < 2; ++i)
#pragma unroll
        for (int j = 0; j < 4; ++j)
            acc[i][j] = (v8f){0.f, 0.f, 0.f, 0.f, 0.f, 0.f, 0.f, 0.f};

    float4 aR[8];   // prefetched x blocks (use .x/.z of each 8-float block)
    float4 bR[4];   // prefetched values (all lanes used)

    // ---- global -> register stage (gather fused for A) --------------------
    auto loadStage = [&](int ks) {
        // k = sh*16 + i ; keep(ks*32+k) = ks*128 + (k>>1)*8 + (k&1)*2
        const float4* ap = (const float4*)(xrow + ks * 128 + sh * 64);
#pragma unroll
        for (int jj = 0; jj < 8; ++jj)
            aR[jj] = ap[jj * 2];
        const float4* bp = (const float4*)(vrow + ks * BK + sh * 16);
#pragma unroll
        for (int q = 0; q < 4; ++q)
            bR[q] = bp[q];
    };

    // ---- register -> LDS stage (convert to packed bf16) -------------------
    auto storeStage = [&](int buf) {
        unsigned int pa[8], pb[8];
#pragma unroll
        for (int jj = 0; jj < 8; ++jj)
            pa[jj] = pack2_bf16(aR[jj].x, aR[jj].z);
#pragma unroll
        for (int q = 0; q < 4; ++q) {
            pb[2 * q + 0] = pack2_bf16(bR[q].x, bR[q].y);
            pb[2 * q + 1] = pack2_bf16(bR[q].z, bR[q].w);
        }
        uint4* da = (uint4*)&sA[buf][sr * LDA + sh * 16];
        da[0] = make_uint4(pa[0], pa[1], pa[2], pa[3]);
        da[1] = make_uint4(pa[4], pa[5], pa[6], pa[7]);
        uint4* db = (uint4*)&sB[buf][sr * LDA + sh * 16];
        db[0] = make_uint4(pb[0], pb[1], pb[2], pb[3]);
        db[1] = make_uint4(pb[4], pb[5], pb[6], pb[7]);
    };

    // ---- WMMA compute for one K-slab --------------------------------------
    auto compute = [&](int buf) {
        Frag a[2];
        Frag b[4];
        // Load ALL fragments first: with the 256-VGPR budget they stay in
        // distinct registers, the 12 ds_load_b128s pipeline, and waits are
        // partial instead of s_wait_dscnt 0 per WMMA pair.
#pragma unroll
        for (int tm = 0; tm < 2; ++tm) {
            // 16-bit A 16x32 layout: lane<16 -> K 0..7 / 16..23, lane>=16 -> +8
            const unsigned short* p =
                &sA[buf][(wm * 32 + tm * 16 + lrow) * LDA + lhi * 8];
            a[tm].q[0] = *(const uint4*)(p);
            a[tm].q[1] = *(const uint4*)(p + 16);
        }
#pragma unroll
        for (int tn = 0; tn < 4; ++tn) {
            // B 32x16 layout: lane<16 -> K 0..7 / 16..23, lane>=16 -> +8
            const unsigned short* p =
                &sB[buf][(wn * 64 + tn * 16 + lrow) * LDA + lhi * 8];
            b[tn].q[0] = *(const uint4*)(p);
            b[tn].q[1] = *(const uint4*)(p + 16);
        }
#pragma unroll
        for (int tn = 0; tn < 4; ++tn) {
#pragma unroll
            for (int tm = 0; tm < 2; ++tm) {
                acc[tm][tn] = __builtin_amdgcn_wmma_f32_16x16x32_bf16(
                    /*neg_a=*/false, a[tm].v,
                    /*neg_b=*/false, b[tn].v,
                    /*c_mod=*/(short)0, acc[tm][tn],
                    /*reuse_a=*/false, /*reuse_b=*/false);
            }
        }
    };

    // ---- pipelined main loop ----------------------------------------------
    loadStage(0);
    storeStage(0);
    __syncthreads();

    for (int ks = 0; ks < NSTEP; ++ks) {
        if (ks + 1 < NSTEP) loadStage(ks + 1);   // overlap global with WMMA
        compute(ks & 1);
        if (ks + 1 < NSTEP) {
            storeStage((ks + 1) & 1);
            __syncthreads();
        }
    }

    // ---- epilogue: bias add + f32 store -----------------------------------
#pragma unroll
    for (int tn = 0; tn < 4; ++tn) {
        const int n  = n0 + wn * 64 + tn * 16 + lrow;
        const float bv = bias[n];
#pragma unroll
        for (int tm = 0; tm < 2; ++tm) {
            const int rbase = m0 + wm * 32 + tm * 16 + lhi * 8;
#pragma unroll
            for (int j = 0; j < 8; ++j) {
                out[(size_t)(rbase + j) * D_OUT + n] = acc[tm][tn][j] + bv;
            }
        }
    }
}

extern "C" void kernel_launch(void* const* d_in, const int* in_sizes, int n_in,
                              void* d_out, int out_size, void* d_ws, size_t ws_size,
                              hipStream_t stream) {
    const float* x      = (const float*)d_in[0];
    const float* values = (const float*)d_in[1];
    const float* bias   = (const float*)d_in[2];
    float* out          = (float*)d_out;

    dim3 grid(D_OUT / BN, M_TOT / BM);   // (32, 128)
    dim3 block(256);
    srnm_spmm_wmma_bf16<<<grid, block, 0, stream>>>(x, values, bias, out);
}
